// EGNNModule_4209067950085
// MI455X (gfx1250) — matrix operand order repaired and checked by reading, not compiled, not run on hardware
//
#include <hip/hip_runtime.h>
#include <hip/hip_bf16.h>
#include <cstddef>
#include <cstdint>

// ---------------------------------------------------------------------------
// EGNN layer for MI455X (gfx1250, wave32, WMMA).
// Edge MLP (156 GFLOP) and node MLP run on v_wmma_f32_16x16x32_f16 with f32
// accumulation. Weights are pre-packed into per-lane WMMA B-fragment layout
// (one contiguous 32B load per lane per fragment).
//
// Edge kernel: 4 nodes (64 edge rows) per block to amortize weight-fragment
// traffic 4x, with hidden-dim chunking (4 chunks x 288 cols) fused into the
// second GEMM so LDS stays under 64KB. B fragments are loaded into distinct
// bf[] registers (one wait-drain per k-step, wmmas issue back-to-back).
// ---------------------------------------------------------------------------

typedef __attribute__((ext_vector_type(16))) _Float16 v16h;
typedef __attribute__((ext_vector_type(8)))  _Float16 h8;
typedef __attribute__((ext_vector_type(8)))  float    v8f;
typedef __attribute__((ext_vector_type(4)))  float    v4f;

#define BATCH 4
#define NN    2048
#define DIM   256
#define KNN   16
#define EDIM  2
#define MDIM  64
#define EIN   515          // 2*DIM + 1 + EDIM
#define HID1  1030         // 2*EIN
#define KS1   17           // 544/32 k-steps over edge_input
#define NT1P  72           // hidden tiles padded (72*16 = 1152)
#define NT2   4            // 64/16
#define KS2T  33           // total GEMM2 k-steps (1056/32)

#define G     4            // nodes per edge block
#define ER    (G*16)       // 64 edge rows per block
#define CHT   18           // N-tiles per hidden chunk (72/4)
#define NCH   4            // hidden chunks
#define CHW   (CHT*16)     // 288 cols per chunk
#define KS2C  9            // GEMM2 k-steps per chunk (288/32)

#define NIN   320          // DIM + MDIM
#define KSN1  10           // 320/32
#define NH    512
#define NTN1  32           // 512/16
#define KSN2  16           // 512/32
#define NTN2  16           // 256/16
#define GN    2            // nodes-rows group for node kernel (32 rows)

__device__ __forceinline__ float silu_f(float v) {
    return v * (1.0f / (1.0f + __expf(-v)));
}

// Load one 16x32 f16 A-fragment from an LDS tile (row-major, rowStride halves).
// ISA 16-bit A layout: lane<16 -> M=lane, K = {kb..kb+7, kb+16..kb+23};
// lane>=16 -> M=lane-16, K base shifted by 8. Two aligned 16B chunks per lane.
__device__ __forceinline__ v16h load_a_frag(const _Float16* tile, int rowStride,
                                            int m, int kbase) {
    const h8 lo = *(const h8*)(tile + m * rowStride + kbase);
    const h8 hi = *(const h8*)(tile + m * rowStride + kbase + 16);
    v16h a;
#pragma unroll
    for (int t = 0; t < 8; ++t) { a[t] = lo[t]; a[t + 8] = hi[t]; }
    return a;
}

__device__ __forceinline__ v8f wmma_f16(v16h a, v16h b, v8f c) {
    return __builtin_amdgcn_wmma_f32_16x16x32_f16(
        false, a, false, b, (short)0, c, false, false);
}

// ---------------------------------------------------------------------------
// Pack fp32 weight [K,N] row-major into WMMA B-fragment order:
// out[((ks*ntiles + nt)*32 + lane)*16 + t] = W[ks*32 + (lane>>4)*8 + (t&7) + (t>>3)*16,
//                                              nt*16 + (lane&15)]  (0 if OOB)
// ---------------------------------------------------------------------------
__global__ void pack_b_kernel(const float* __restrict__ W, int K, int N,
                              _Float16* __restrict__ out, int ksteps, int ntiles) {
    int total = ksteps * ntiles * 512;
    for (int t = blockIdx.x * 256 + threadIdx.x; t < total; t += gridDim.x * 256) {
        int h    = t & 15;
        int lane = (t >> 4) & 31;
        int tile = t >> 9;
        int nt   = tile % ntiles;
        int ks   = tile / ntiles;
        int n    = nt * 16 + (lane & 15);
        int kk   = ks * 32 + (lane >> 4) * 8 + (h & 7) + (h >> 3) * 16;
        out[t] = (kk < K && n < N) ? (_Float16)W[(size_t)kk * N + n]
                                   : (_Float16)0.0f;
    }
}

// ---------------------------------------------------------------------------
// kNN: one block per (b,i). Ranking semantics match the reference:
// masked pairs -> 1e5, self -> -1, adjacent(non-self) -> 0; then 16 smallest
// with lowest-index tie break (top_k stability).
// ---------------------------------------------------------------------------
__global__ __launch_bounds__(256) void knn_kernel(
    const float* __restrict__ coors, const unsigned char* __restrict__ adj,
    const unsigned char* __restrict__ mask, int* __restrict__ idx_out,
    float* __restrict__ relk_out, float* __restrict__ rowmask_out) {
    __shared__ float rank[NN];
    __shared__ float bv[256];
    __shared__ int   bi[256];
    __shared__ int   selI[KNN];
    __shared__ float selV[KNN];

    const int row = blockIdx.x;           // b*N + i
    const int b   = row >> 11;
    const int i   = row & (NN - 1);
    const int tid = threadIdx.x;

    const float cx = coors[(size_t)row * 3 + 0];
    const float cy = coors[(size_t)row * 3 + 1];
    const float cz = coors[(size_t)row * 3 + 2];
    const int mi = mask[row];
    const float*         cb      = coors + (size_t)b * NN * 3;
    const unsigned char* adjRow  = adj + (size_t)row * NN;
    const unsigned char* maskRow = mask + (size_t)b * NN;

    for (int j = tid; j < NN; j += 256) {
        float dx = cx - cb[j * 3 + 0];
        float dy = cy - cb[j * 3 + 1];
        float dz = cz - cb[j * 3 + 2];
        float d  = dx * dx + dy * dy + dz * dz;
        float r  = (mi && maskRow[j]) ? d : 1.0e5f;
        if (j == i)         r = -1.0f;
        else if (adjRow[j]) r = 0.0f;
        rank[j] = r;
    }
    __syncthreads();

    for (int t = 0; t < KNN; ++t) {
        float bvv = 3.0e38f;
        int   bii = 1 << 30;
        for (int j = tid; j < NN; j += 256) {     // ascending j: ties keep low j
            float v = rank[j];
            if (v < bvv) { bvv = v; bii = j; }
        }
        bv[tid] = bvv; bi[tid] = bii;
        __syncthreads();
        for (int off = 128; off > 0; off >>= 1) {
            if (tid < off) {
                float v2 = bv[tid + off]; int i2 = bi[tid + off];
                if (v2 < bv[tid] || (v2 == bv[tid] && i2 < bi[tid])) {
                    bv[tid] = v2; bi[tid] = i2;
                }
            }
            __syncthreads();
        }
        if (tid == 0) { selI[t] = bi[0]; selV[t] = bv[0]; rank[bi[0]] = 3.0e38f; }
        __syncthreads();
    }

    if (tid < KNN) {
        int j = selI[tid];
        float dx = cx - cb[j * 3 + 0];
        float dy = cy - cb[j * 3 + 1];
        float dz = cz - cb[j * 3 + 2];
        float d  = dx * dx + dy * dy + dz * dz;   // raw rel_dist (pre-masking)
        bool pair = (mi != 0) && (maskRow[j] != 0) && (selV[tid] <= 0.0f);
        idx_out[row * KNN + tid]     = j;
        relk_out[row * KNN + tid]    = d;
        rowmask_out[row * KNN + tid] = pair ? 1.0f : 0.0f;
    }
}

// ---------------------------------------------------------------------------
// Build one 64-row x 32-col k-slice of edge_input (f16) into LDS.
// Columns: [feats_i(0..255) | feats_j(256..511) | rel_dist(512) |
//           edge(513..514) | zero pad]
// ---------------------------------------------------------------------------
__device__ __forceinline__ void build_a_slice(
    _Float16* abuf, int ks, int g0, const int* idxS, const float* relS,
    const float* __restrict__ emb, const float* __restrict__ embB,
    const float* __restrict__ feat, int tid) {
#pragma unroll
    for (int it = 0; it < ER * 32 / 256; ++it) {
        int e  = tid + it * 256;
        int er = e >> 5;                 // edge row 0..63
        int cc = e & 31;
        int col = ks * 32 + cc;
        int nd = er >> 4;                // node within group
        int g  = g0 + nd;                // global node row (b*N+i)
        float v;
        if (col < DIM)           v = emb[(size_t)g * DIM + col];
        else if (col < 2 * DIM)  v = embB[(size_t)idxS[er] * DIM + (col - DIM)];
        else if (col == 2 * DIM) v = relS[er];
        else if (col < EIN)      v = feat[((size_t)g * NN + idxS[er]) * EDIM + (col - 513)];
        else                     v = 0.0f;
        abuf[er * 32 + cc] = (_Float16)v;
    }
}

// ---------------------------------------------------------------------------
// Edge MLP: 4 nodes (64 edge rows = 4 M-tiles) per block.
// Per hidden chunk c (288 cols): GEMM1 (17 ks, A double-buffered 64x32 slice)
// -> SiLU -> h-chunk in LDS -> GEMM2 partial accumulate (9 ks) into
// persistent registers. Final: bias, SiLU, row mask, sum over 16 edges.
// ---------------------------------------------------------------------------
__global__ __launch_bounds__(256) void edge_mlp_kernel(
    const float* __restrict__ emb, const float* __restrict__ feat,
    const int* __restrict__ idx, const float* __restrict__ relk,
    const float* __restrict__ rowmask,
    const _Float16* __restrict__ W1p, const float* __restrict__ eb1,
    const _Float16* __restrict__ W2p, const float* __restrict__ eb2,
    float* __restrict__ m_i) {
    __shared__ _Float16 aBuf[2][ER * 32];    // 2 x 4 KB A k-slices
    __shared__ _Float16 hC[ER * CHW];        // 36 KB hidden chunk (f16)
    __shared__ int   idxS[ER];
    __shared__ float rmS[ER];
    __shared__ float relS[ER];
    __shared__ float sum2[G][2][MDIM];       // 2 KB

    const int tid = threadIdx.x;
    const int g0  = blockIdx.x * G;          // first node row (b*N+i); same b
    const int b   = g0 >> 11;

    if (tid < ER) {
        idxS[tid] = idx[g0 * KNN + tid];
        rmS[tid]  = rowmask[g0 * KNN + tid];
        relS[tid] = relk[g0 * KNN + tid];
    }
    __syncthreads();

    const int w    = tid >> 5;
    const int lane = tid & 31;
    const int mlo  = lane & 15;
    const int koff = (lane >> 4) * 8;
    const int nlo  = lane & 15;
    const int mh   = lane >> 4;
    const int mt   = w >> 1;                 // M-tile (node) for this wave
    const int nt0  = (w & 1) * 9;            // first of 9 chunk-local N-tiles

    const float* embB = emb + (size_t)b * NN * DIM;

    v8f acc2[2];                             // persistent GEMM2 accumulators
    acc2[0] = {}; acc2[1] = {};

    for (int c = 0; c < NCH; ++c) {
        v8f acc[9];
#pragma unroll
        for (int s = 0; s < 9; ++s) acc[s] = {};

        build_a_slice(aBuf[0], 0, g0, idxS, relS, emb, embB, feat, tid);
        __syncthreads();

        for (int ks = 0; ks < KS1; ++ks) {
            const _Float16* ab = aBuf[ks & 1];
            v16h a = load_a_frag(ab, 32, mt * 16 + mlo, koff);
            const _Float16* bp =
                W1p + ((size_t)(ks * NT1P + c * CHT + nt0) * 32 + lane) * 16;
            v16h bf[9];
#pragma unroll
            for (int s = 0; s < 9; ++s)      // one load clause, one drain
                bf[s] = *(const v16h*)(bp + (size_t)s * 512);
#pragma unroll
            for (int s = 0; s < 9; ++s)
                acc[s] = wmma_f16(a, bf[s], acc[s]);
            if (ks < KS1 - 1)
                build_a_slice(aBuf[(ks + 1) & 1], ks + 1, g0, idxS, relS,
                              emb, embB, feat, tid);
            __syncthreads();
        }

        // SiLU epilogue -> h chunk (exact zeros for padded hidden cols)
#pragma unroll
        for (int s = 0; s < 9; ++s) {
            int ntRel = nt0 + s;
            int gcol  = (c * CHT + ntRel) * 16 + nlo;   // global hidden col
            int lcol  = ntRel * 16 + nlo;               // chunk-local col
            float bias = (gcol < HID1) ? eb1[gcol] : 0.0f;
#pragma unroll
            for (int r = 0; r < 8; ++r) {
                int M = mt * 16 + mh * 8 + r;
                float v = silu_f(acc[s][r] + bias);
                hC[M * CHW + lcol] = (gcol < HID1) ? (_Float16)v : (_Float16)0.0f;
            }
        }
        __syncthreads();

        // GEMM2 partial over this chunk's 288 hidden cols
        {
            const int ks0 = c * KS2C;
#pragma unroll
            for (int kk = 0; kk < KS2C; ++kk) {
                v16h a = load_a_frag(hC, CHW, mt * 16 + mlo, kk * 32 + koff);
                const _Float16* bp2 =
                    W2p + ((size_t)((ks0 + kk) * NT2 + (w & 1) * 2) * 32 + lane) * 16;
                v16h b0 = *(const v16h*)bp2;
                v16h b1 = *(const v16h*)(bp2 + 512);
                acc2[0] = wmma_f16(a, b0, acc2[0]);
                acc2[1] = wmma_f16(a, b1, acc2[1]);
            }
        }
        __syncthreads();                     // hC/aBuf reused next chunk
    }

    // Final: bias, SiLU, pair-mask, sum over the node's 16 edge rows
#pragma unroll
    for (int s = 0; s < 2; ++s) {
        int col = ((w & 1) * 2 + s) * 16 + nlo;          // 0..63
        float bias = eb2[col];
        float sum = 0.0f;
#pragma unroll
        for (int r = 0; r < 8; ++r) {
            int Ml = mh * 8 + r;                         // row within node tile
            sum += silu_f(acc2[s][r] + bias) * rmS[mt * 16 + Ml];
        }
        sum2[mt][mh][col] = sum;
    }
    __syncthreads();
    {
        int nd = tid >> 6, col = tid & 63;               // 4 nodes x 64 cols
        m_i[(size_t)(g0 + nd) * MDIM + col] = sum2[nd][0][col] + sum2[nd][1][col];
    }
}

// ---------------------------------------------------------------------------
// Node MLP: 32 rows (2 M-tiles) per block.
// out = silu([emb|m_i] @ nW1 + nb1) @ nW2 + nb2 + emb
// ---------------------------------------------------------------------------
__global__ __launch_bounds__(256) void node_mlp_kernel(
    const float* __restrict__ emb, const float* __restrict__ m_i,
    const _Float16* __restrict__ W1p, const float* __restrict__ nb1,
    const _Float16* __restrict__ W2p, const float* __restrict__ nb2,
    float* __restrict__ out) {
    __shared__ _Float16 aT[GN * 16 * NIN];   // 20 KB
    __shared__ _Float16 hT[GN * 16 * NH];    // 32 KB

    const int tid = threadIdx.x;
    const int r0  = blockIdx.x * (GN * 16);

    for (int t = tid; t < GN * 16 * NIN; t += 256) {
        int rr = t / NIN, c = t - rr * NIN;
        size_t g = (size_t)(r0 + rr);
        float v = (c < DIM) ? emb[g * DIM + c] : m_i[g * MDIM + (c - DIM)];
        aT[t] = (_Float16)v;
    }
    __syncthreads();

    const int w    = tid >> 5;
    const int lane = tid & 31;
    const int mlo  = lane & 15;
    const int koff = (lane >> 4) * 8;
    const int nlo  = lane & 15;
    const int mh   = lane >> 4;
    const int mt   = w >> 2;                 // M-tile, 4 waves per tile

    // GEMM1: 2 mt x 32 nt = 64 pairs -> 8 per wave; K = 320
    v8f acc[8];
#pragma unroll
    for (int s = 0; s < 8; ++s) acc[s] = {};
    for (int ks = 0; ks < KSN1; ++ks) {
        v16h a = load_a_frag(aT, NIN, mt * 16 + mlo, ks * 32 + koff);
        const _Float16* bp =
            W1p + ((size_t)(ks * NTN1 + (w & 3) * 8) * 32 + lane) * 16;
        v16h bf[8];
#pragma unroll
        for (int s = 0; s < 8; ++s)
            bf[s] = *(const v16h*)(bp + (size_t)s * 512);
#pragma unroll
        for (int s = 0; s < 8; ++s)
            acc[s] = wmma_f16(a, bf[s], acc[s]);
    }
#pragma unroll
    for (int s = 0; s < 8; ++s) {
        int col = ((w & 3) * 8 + s) * 16 + nlo;
        float bias = nb1[col];
#pragma unroll
        for (int r = 0; r < 8; ++r) {
            int M = mt * 16 + mh * 8 + r;
            hT[M * NH + col] = (_Float16)silu_f(acc[s][r] + bias);
        }
    }
    __syncthreads();

    // GEMM2: 2 mt x 16 nt = 32 pairs -> 4 per wave; K = 512; +bias +residual
    v8f acc2[4];
#pragma unroll
    for (int s = 0; s < 4; ++s) acc2[s] = {};
    for (int ks = 0; ks < KSN2; ++ks) {
        v16h a = load_a_frag(hT, NH, mt * 16 + mlo, ks * 32 + koff);
        const _Float16* bp =
            W2p + ((size_t)(ks * NTN2 + (w & 3) * 4) * 32 + lane) * 16;
        v16h bf[4];
#pragma unroll
        for (int s = 0; s < 4; ++s)
            bf[s] = *(const v16h*)(bp + (size_t)s * 512);
#pragma unroll
        for (int s = 0; s < 4; ++s)
            acc2[s] = wmma_f16(a, bf[s], acc2[s]);
    }
#pragma unroll
    for (int s = 0; s < 4; ++s) {
        int col = ((w & 3) * 4 + s) * 16 + nlo;
        float bias = nb2[col];
#pragma unroll
        for (int r = 0; r < 8; ++r) {
            int M = mt * 16 + mh * 8 + r;
            size_t g = (size_t)(r0 + M);
            out[g * DIM + col] = acc2[s][r] + bias + emb[g * DIM + col];
        }
    }
}

// ---------------------------------------------------------------------------
// Pass-through copies for the tuple tail. Non-temporal b128 streams so the
// 200 MB copy does not evict packed weights / emb from L2.
// ---------------------------------------------------------------------------
__global__ void copy_f32_kernel(const float* __restrict__ s, float* __restrict__ d,
                                size_t n) {
    size_t n4 = n >> 2;
    for (size_t t = (size_t)blockIdx.x * 256 + threadIdx.x; t < n4;
         t += (size_t)gridDim.x * 256) {
        v4f v = __builtin_nontemporal_load((const v4f*)s + t);
        __builtin_nontemporal_store(v, (v4f*)d + t);
    }
    if (blockIdx.x == 0 && threadIdx.x < (n & 3)) {
        size_t i = (n4 << 2) + threadIdx.x;
        d[i] = s[i];
    }
}
__global__ void copy_u8_f32_kernel(const unsigned char* __restrict__ s,
                                   float* __restrict__ d, size_t n) {
    size_t n4 = n >> 2;
    for (size_t t = (size_t)blockIdx.x * 256 + threadIdx.x; t < n4;
         t += (size_t)gridDim.x * 256) {
        unsigned int v = __builtin_nontemporal_load((const unsigned int*)s + t);
        v4f o;
        o[0] = (v & 0x000000ffu) ? 1.0f : 0.0f;
        o[1] = (v & 0x0000ff00u) ? 1.0f : 0.0f;
        o[2] = (v & 0x00ff0000u) ? 1.0f : 0.0f;
        o[3] = (v & 0xff000000u) ? 1.0f : 0.0f;
        __builtin_nontemporal_store(o, (v4f*)d + t);
    }
    if (blockIdx.x == 0 && threadIdx.x < (n & 3)) {
        size_t i = (n4 << 2) + threadIdx.x;
        d[i] = s[i] ? 1.0f : 0.0f;
    }
}

// ---------------------------------------------------------------------------
extern "C" void kernel_launch(void* const* d_in, const int* in_sizes, int n_in,
                              void* d_out, int out_size, void* d_ws, size_t ws_size,
                              hipStream_t stream) {
    const float*         emb   = (const float*)d_in[0];
    const float*         coors = (const float*)d_in[1];
    const unsigned char* adj   = (const unsigned char*)d_in[2];   // bool
    const float*         feat  = (const float*)d_in[3];
    const unsigned char* mask  = (const unsigned char*)d_in[4];   // bool
    const float* eW1 = (const float*)d_in[5];
    const float* eb1 = (const float*)d_in[6];
    const float* eW2 = (const float*)d_in[7];
    const float* eb2 = (const float*)d_in[8];
    const float* nW1 = (const float*)d_in[9];
    const float* nb1 = (const float*)d_in[10];
    const float* nW2 = (const float*)d_in[11];
    const float* nb2 = (const float*)d_in[12];

    const int rows = BATCH * NN;          // 8192

    // ---- workspace carve (256B aligned) ----
    char* ws = (char*)d_ws;
    size_t off = 0;
    auto carve = [&](size_t bytes) {
        void* p = ws + off;
        off += (bytes + 255) & ~(size_t)255;
        return p;
    };
    _Float16* eW1p = (_Float16*)carve((size_t)KS1 * NT1P * 512 * 2);  // 1.25 MB
    _Float16* eW2p = (_Float16*)carve((size_t)KS2T * NT2 * 512 * 2);  // 135 KB
    _Float16* nW1p = (_Float16*)carve((size_t)KSN1 * NTN1 * 512 * 2); // 328 KB
    _Float16* nW2p = (_Float16*)carve((size_t)KSN2 * NTN2 * 512 * 2); // 262 KB
    int*   idxW  = (int*)carve((size_t)rows * KNN * 4);
    float* relkW = (float*)carve((size_t)rows * KNN * 4);
    float* rmW   = (float*)carve((size_t)rows * KNN * 4);
    float* miW   = (float*)carve((size_t)rows * MDIM * 4);
    (void)ws_size;

    // ---- weight packing (f32 -> f16 WMMA B-fragment layout) ----
    {
        int t1 = KS1 * NT1P * 512, t2 = KS2T * NT2 * 512;
        int t3 = KSN1 * NTN1 * 512, t4 = KSN2 * NTN2 * 512;
        pack_b_kernel<<<(t1 + 255) / 256, 256, 0, stream>>>(eW1, EIN,  HID1, eW1p, KS1,  NT1P);
        pack_b_kernel<<<(t2 + 255) / 256, 256, 0, stream>>>(eW2, HID1, MDIM, eW2p, KS2T, NT2);
        pack_b_kernel<<<(t3 + 255) / 256, 256, 0, stream>>>(nW1, NIN,  NH,   nW1p, KSN1, NTN1);
        pack_b_kernel<<<(t4 + 255) / 256, 256, 0, stream>>>(nW2, NH,   DIM,  nW2p, KSN2, NTN2);
    }

    // ---- kNN selection ----
    knn_kernel<<<rows, 256, 0, stream>>>(coors, adj, mask, idxW, relkW, rmW);

    // ---- edge MLP + masked K-reduction ----
    edge_mlp_kernel<<<rows / G, 256, 0, stream>>>(emb, feat, idxW, relkW, rmW,
                                                  eW1p, eb1, eW2p, eb2, miW);

    // ---- node MLP + residual -> node_out ----
    float* out = (float*)d_out;
    node_mlp_kernel<<<rows / (GN * 16), 256, 0, stream>>>(emb, miW, nW1p, nb1,
                                                          nW2p, nb2, out);

    // ---- tuple tail: coors, adj, feat, mask (bool -> float) ----
    size_t o = (size_t)in_sizes[0];                      // node_out section
    copy_f32_kernel<<<1024, 256, 0, stream>>>(coors, out + o, (size_t)in_sizes[1]);
    o += (size_t)in_sizes[1];
    copy_u8_f32_kernel<<<4096, 256, 0, stream>>>(adj, out + o, (size_t)in_sizes[2]);
    o += (size_t)in_sizes[2];
    copy_f32_kernel<<<8192, 256, 0, stream>>>(feat, out + o, (size_t)in_sizes[3]);
    o += (size_t)in_sizes[3];
    copy_u8_f32_kernel<<<32, 256, 0, stream>>>(mask, out + o, (size_t)in_sizes[4]);
}
